// StripedHyenaBlock_81303730913668
// MI455X (gfx1250) — compile-verified, matrix operand order checked by reading, not run on hardware
//
#include <hip/hip_runtime.h>
#include <math.h>

typedef __attribute__((ext_vector_type(16))) _Float16 v16h;
typedef __attribute__((ext_vector_type(8)))  float    v8f;

union FragH { v16h h; unsigned int u[8]; };
union H8   { uint4 u; _Float16 h[8]; };

__device__ __forceinline__ float gelu_f(float x) {
    return 0.5f * x * (1.0f + erff(x * 0.70710678118654752f));
}

// K-index mapping for 16-bit A/B WMMA fragments (CDNA5 ISA §7.12.2)
__device__ __forceinline__ int frag_kk(int v, int lh) {
    return ((v < 4) ? (2 * v) : (16 + 2 * (v - 4))) + lh * 8;
}

// ---------------------------------------------------------------------------
// Tensor Data Mover (TDM) 2D f16 tile load: global -> LDS (packed rows)
// 6-arg builtin form: (u32x4 g0, i32x8 g1, i32x4 g2, i32x4 g3, i32x8 pad, i32 cpol)
// ---------------------------------------------------------------------------
#if defined(__HIP_DEVICE_COMPILE__) && defined(__gfx1250__) && \
    __has_builtin(__builtin_amdgcn_tensor_load_to_lds) && \
    __has_builtin(__builtin_amdgcn_s_wait_tensorcnt)
#define HAVE_TDM 1
typedef unsigned int u32x4 __attribute__((ext_vector_type(4)));
typedef int          i32x8 __attribute__((ext_vector_type(8)));
typedef int          i32x4 __attribute__((ext_vector_type(4)));

__device__ __forceinline__ void tdm_load_2d_f16(unsigned lds_addr, const _Float16* gptr,
                                                unsigned tile_k, unsigned tile_rows,
                                                unsigned dim0_rem, unsigned dim1_rem,
                                                unsigned stride_elems)
{
    unsigned long long ga = (unsigned long long)(size_t)gptr;
    u32x4 g0;
    g0[0] = 1u;                                             // count=1, user mode
    g0[1] = lds_addr;                                       // LDS byte address
    g0[2] = (unsigned)(ga & 0xFFFFFFFFu);                   // global_addr[31:0]
    g0[3] = (unsigned)((ga >> 32) & 0x01FFFFFFu) | (2u << 30); // addr[56:32], type=2
    i32x8 g1;
    g1[0] = (int)(1u << 16);                                // data_size=1 (2 bytes)
    g1[1] = (int)((dim0_rem & 0xFFFFu) << 16);              // tensor_dim0[15:0]
    g1[2] = (int)(((dim0_rem >> 16) & 0xFFFFu) | ((dim1_rem & 0xFFFFu) << 16));
    g1[3] = (int)(((dim1_rem >> 16) & 0xFFFFu) | ((tile_k & 0xFFFFu) << 16)); // tile_dim0
    g1[4] = (int)(tile_rows & 0xFFFFu);                     // tile_dim1 (tile_dim2=0)
    g1[5] = (int)stride_elems;                              // tensor_dim0_stride[31:0]
    g1[6] = 0;
    g1[7] = 0;
    i32x4 z4 = {0, 0, 0, 0};
    i32x8 z8 = {0, 0, 0, 0, 0, 0, 0, 0};
    __builtin_amdgcn_tensor_load_to_lds(g0, g1, z4, z4, z8, 0);
}
#endif

// ---------------------------------------------------------------------------
// Weight convert+transpose: Wt[n][k] = f16(W[k][n]), zero-padded to Npad rows
// ---------------------------------------------------------------------------
__launch_bounds__(256)
__global__ void wconvert_kernel(const float* __restrict__ W, _Float16* __restrict__ Wt,
                                int K, int N, int Npad)
{
    size_t idx = (size_t)blockIdx.x * 256 + threadIdx.x;
    size_t total = (size_t)Npad * K;
    if (idx >= total) return;
    int n = (int)(idx / K);
    int k = (int)(idx % K);
    Wt[idx] = (n < N) ? (_Float16)W[(size_t)k * N + n] : (_Float16)0.0f;
}

__launch_bounds__(256)
__global__ void aconvert_kernel(const float* __restrict__ A, _Float16* __restrict__ A16,
                                size_t total)
{
    size_t idx = (size_t)blockIdx.x * 256 + threadIdx.x;
    if (idx < total) A16[idx] = (_Float16)A[idx];
}

// ---------------------------------------------------------------------------
// WMMA GEMM: C = act(A16[M,K] @ Bt16[N,K]^T + bias) (+resid). f16 in, f32 acc.
// Double-buffered TDM staging (DMA for stage i+1 overlaps WMMA for stage i).
// ---------------------------------------------------------------------------
#define BM 128
#define BN 128
#define BK 64

__launch_bounds__(256)
__global__ void gemm_wmma_kernel(const _Float16* __restrict__ A,
                                 const _Float16* __restrict__ Bt,
                                 const float* __restrict__ bias,
                                 const float* __restrict__ resid,
                                 float* __restrict__ C,
                                 _Float16* __restrict__ Ch,
                                 int M, int N, int Npad, int K, int act)
{
    __shared__ __align__(16) _Float16 As[2][BM * BK];   // [row][k]
    __shared__ __align__(16) _Float16 Bs[2][BN * BK];   // [n][k]

    const int tid  = threadIdx.x;
    const int wave = tid >> 5;
    const int lane = tid & 31;
    const int lm   = lane & 15;
    const int lh   = lane >> 4;
    const int wm   = wave & 3;            // 4 waves along M (32 rows)
    const int wn   = wave >> 2;           // 2 waves along N (64 cols)
    const int row0 = blockIdx.y * BM;
    const int col0 = blockIdx.x * BN;
    const int T    = K / BK;

    v8f acc[2][4];
    #pragma unroll
    for (int t = 0; t < 2; ++t)
        #pragma unroll
        for (int j = 0; j < 4; ++j)
            #pragma unroll
            for (int r = 0; r < 8; ++r)
                acc[t][j][r] = 0.0f;

    auto compute_tile = [&](const _Float16* Ap, const _Float16* Bp) {
        #pragma unroll
        for (int kc = 0; kc < BK; kc += 32) {
            FragH fa[2], fb[4];
            #pragma unroll
            for (int t = 0; t < 2; ++t)
                #pragma unroll
                for (int v = 0; v < 8; ++v)
                    fa[t].u[v] = *reinterpret_cast<const unsigned int*>(
                        &Ap[(wm * 32 + t * 16 + lm) * BK + kc + frag_kk(v, lh)]);
            #pragma unroll
            for (int j = 0; j < 4; ++j)
                #pragma unroll
                for (int v = 0; v < 8; ++v)
                    fb[j].u[v] = *reinterpret_cast<const unsigned int*>(
                        &Bp[(wn * 64 + j * 16 + lm) * BK + kc + frag_kk(v, lh)]);
            #pragma unroll
            for (int t = 0; t < 2; ++t)
                #pragma unroll
                for (int j = 0; j < 4; ++j)
                    acc[t][j] = __builtin_amdgcn_wmma_f32_16x16x32_f16(
                        false, fa[t].h, false, fb[j].h, (short)0, acc[t][j], false, false);
        }
    };

#if defined(HAVE_TDM)
    const unsigned lds_as[2] = { (unsigned)(size_t)(void*)&As[0][0],
                                 (unsigned)(size_t)(void*)&As[1][0] };
    const unsigned lds_bs[2] = { (unsigned)(size_t)(void*)&Bs[0][0],
                                 (unsigned)(size_t)(void*)&Bs[1][0] };
    if (wave == 0) {                      // prologue: stage 0 tiles
        tdm_load_2d_f16(lds_as[0], A  + (size_t)row0 * K, BK, BM,
                        (unsigned)K, (unsigned)(M - row0), (unsigned)K);
        tdm_load_2d_f16(lds_bs[0], Bt + (size_t)col0 * K, BK, BN,
                        (unsigned)K, (unsigned)(Npad - col0), (unsigned)K);
    }
    for (int i = 0; i < T; ++i) {
        const int pb = i & 1;
        if (wave == 0) {
            if (i + 1 < T) {              // overlap next stage DMA with compute
                const int k1 = (i + 1) * BK;
                tdm_load_2d_f16(lds_as[pb ^ 1], A  + (size_t)row0 * K + k1, BK, BM,
                                (unsigned)(K - k1), (unsigned)(M - row0), (unsigned)K);
                tdm_load_2d_f16(lds_bs[pb ^ 1], Bt + (size_t)col0 * K + k1, BK, BN,
                                (unsigned)(K - k1), (unsigned)(Npad - col0), (unsigned)K);
                __builtin_amdgcn_s_wait_tensorcnt(2);   // current stage complete
            } else {
                __builtin_amdgcn_s_wait_tensorcnt(0);
            }
        }
        __syncthreads();                  // current buffers visible to all waves
        compute_tile(&As[pb][0], &Bs[pb][0]);
        asm volatile("s_wait_dscnt 0x0" ::: "memory");   // drain LDS reads
        __syncthreads();                  // safe to DMA over this buffer next+1
    }
#else
    for (int i = 0; i < T; ++i) {
        const int k0 = i * BK;
        #pragma unroll
        for (int c = 0; c < 4; ++c) {     // A tile: 1024 x 16B chunks
            int idx = tid + c * 256;
            int r = idx >> 3, c8 = (idx & 7) * 8;
            *reinterpret_cast<uint4*>(&As[0][r * BK + c8]) =
                *reinterpret_cast<const uint4*>(&A[(size_t)(row0 + r) * K + k0 + c8]);
        }
        #pragma unroll
        for (int c = 0; c < 4; ++c) {     // B tile (padded, no guard)
            int idx = tid + c * 256;
            int n = idx >> 3, c8 = (idx & 7) * 8;
            *reinterpret_cast<uint4*>(&Bs[0][n * BK + c8]) =
                *reinterpret_cast<const uint4*>(&Bt[(size_t)(col0 + n) * K + k0 + c8]);
        }
        __syncthreads();
        compute_tile(&As[0][0], &Bs[0][0]);
        __syncthreads();
    }
#endif

    // epilogue: C layout — VGPR r, lane lm => element (m = r + 8*lh, n = lm)
    #pragma unroll
    for (int t = 0; t < 2; ++t)
        #pragma unroll
        for (int j = 0; j < 4; ++j)
            #pragma unroll
            for (int r = 0; r < 8; ++r) {
                int m  = wm * 32 + t * 16 + r + lh * 8;
                int gn = col0 + wn * 64 + j * 16 + lm;
                if (gn < N) {
                    float val = acc[t][j][r];
                    if (bias)  val += bias[gn];
                    if (act)   val = gelu_f(val);
                    if (resid) val += resid[(size_t)(row0 + m) * N + gn];
                    if (C)  C[(size_t)(row0 + m) * N + gn]  = val;
                    if (Ch) Ch[(size_t)(row0 + m) * N + gn] = (_Float16)val;
                }
            }
}

// ---------------------------------------------------------------------------
// Hyena causal conv (simple, F < 16)
// ---------------------------------------------------------------------------
__launch_bounds__(256)
__global__ void hyena_conv_small_kernel(const float* __restrict__ h_in,
                                        const float* __restrict__ p,
                                        float* __restrict__ h_out,
                                        _Float16* __restrict__ h16,
                                        int L, int D, int F, int order_idx, int orderF,
                                        int write16)
{
    const int b = blockIdx.y;
    const int l = blockIdx.x;
    const int tid = threadIdx.x;
    __shared__ float kf[16];
    if (tid < F)
        kf[tid] = p[((size_t)(b * L + l)) * orderF + order_idx * F + tid];
    __syncthreads();
    const int mmax = (l < F - 1) ? l : (F - 1);
    for (int d = tid; d < D; d += 256) {
        float acc = 0.0f;
        const float* hrow = &h_in[((size_t)(b * L + l)) * D + d];
        for (int m = 0; m <= mmax; ++m)
            acc += kf[m] * hrow[-(size_t)m * D];
        float g = gelu_f(acc);
        h_out[((size_t)(b * L + l)) * D + d] = g;
        if (write16) h16[((size_t)(b * L + l)) * D + d] = (_Float16)g;
    }
}

// ---------------------------------------------------------------------------
// Hyena causal conv (register-tiled, F % 16 == 0): 16 outputs per thread
// ---------------------------------------------------------------------------
#define CTL 16
__launch_bounds__(256)
__global__ void hyena_conv_tiled_kernel(const float* __restrict__ h_in,
                                        const float* __restrict__ p,
                                        float* __restrict__ h_out,
                                        _Float16* __restrict__ h16,
                                        int L, int D, int F, int order_idx, int orderF,
                                        int write16)
{
    const int b  = blockIdx.z;
    const int lb = blockIdx.x * CTL;
    const int d  = blockIdx.y * 256 + threadIdx.x;
    __shared__ float kf[CTL * 256];   // CTL rows x F (F <= 256)

    for (int i = threadIdx.x; i < CTL * F; i += 256) {
        int il = i / F, m = i - il * F;
        kf[il * F + m] = p[((size_t)(b * L + lb + il)) * orderF + order_idx * F + m];
    }
    __syncthreads();

    const float* hcol = h_in + (size_t)b * L * D + d;
    float acc[CTL];
    #pragma unroll
    for (int i = 0; i < CTL; ++i) acc[i] = 0.0f;

    for (int mm = 0; mm < F; mm += CTL) {
        float hv[2 * CTL - 1];
        #pragma unroll
        for (int c = 0; c < 2 * CTL - 1; ++c) {
            int t = lb - mm - (CTL - 1) + c;
            hv[c] = (t >= 0) ? hcol[(size_t)t * D] : 0.0f;
        }
        #pragma unroll
        for (int j = 0; j < CTL; ++j)
            #pragma unroll
            for (int i = 0; i < CTL; ++i)
                acc[i] += kf[i * F + mm + j] * hv[(CTL - 1) + i - j];
    }
    #pragma unroll
    for (int i = 0; i < CTL; ++i) {
        float g = gelu_f(acc[i]);
        h_out[((size_t)(b * L + lb + i)) * D + d] = g;
        if (write16) h16[((size_t)(b * L + lb + i)) * D + d] = (_Float16)g;
    }
}

// ---------------------------------------------------------------------------
// out = LayerNorm(a + b) * g + beta  (+ optional f16 companion)
// ---------------------------------------------------------------------------
__launch_bounds__(256)
__global__ void ln_res_kernel(const float* __restrict__ a,
                              const float* __restrict__ b,
                              const float* __restrict__ g,
                              const float* __restrict__ beta,
                              float* __restrict__ out,
                              _Float16* __restrict__ out16, int D)
{
    const int row = blockIdx.x;
    const int tid = threadIdx.x;
    __shared__ float red[256];
    float loc[3];
    float sum = 0.0f;
    #pragma unroll
    for (int i = 0; i < 3; ++i) {
        int d = tid + i * 256;
        float v = a[(size_t)row * D + d] + b[(size_t)row * D + d];
        loc[i] = v;
        sum += v;
    }
    red[tid] = sum; __syncthreads();
    for (int s = 128; s > 0; s >>= 1) { if (tid < s) red[tid] += red[tid + s]; __syncthreads(); }
    const float mean = red[0] / (float)D;
    __syncthreads();
    float vs = 0.0f;
    #pragma unroll
    for (int i = 0; i < 3; ++i) { float dd = loc[i] - mean; vs += dd * dd; }
    red[tid] = vs; __syncthreads();
    for (int s = 128; s > 0; s >>= 1) { if (tid < s) red[tid] += red[tid + s]; __syncthreads(); }
    const float inv = rsqrtf(red[0] / (float)D + 1e-5f);
    #pragma unroll
    for (int i = 0; i < 3; ++i) {
        int d = tid + i * 256;
        float o = (loc[i] - mean) * inv * g[d] + beta[d];
        out[(size_t)row * D + d] = o;
        if (out16) out16[(size_t)row * D + d] = (_Float16)o;
    }
}

// ---------------------------------------------------------------------------
// RoPE -> f16 output (consumed only by Q/K projection GEMMs)
// ---------------------------------------------------------------------------
__launch_bounds__(256)
__global__ void rope_kernel(const float* __restrict__ x,
                            const float* __restrict__ sn,
                            const float* __restrict__ cs,
                            _Float16* __restrict__ out16, int L, int D)
{
    const int row = blockIdx.x;
    const int l = row % L;
    const int Dh = D / 2;
    for (int d = threadIdx.x; d < Dh; d += 256) {
        float c  = cs[(size_t)l * Dh + d];
        float s  = sn[(size_t)l * Dh + d];
        float x1 = x[(size_t)row * D + d];
        float x2 = x[(size_t)row * D + Dh + d];
        out16[(size_t)row * D + d]      = (_Float16)(x1 * c - x2 * s);
        out16[(size_t)row * D + Dh + d] = (_Float16)(x2 * c + x1 * s);
    }
}

// ---------------------------------------------------------------------------
// Flash attention, one wave / 16-query block. f16 Q,K,V,O. hd=64, scale=1/8.
// K tiles via double-buffered TDM; V prefetched one step ahead in registers.
// ---------------------------------------------------------------------------
__launch_bounds__(32)
__global__ void attn_kernel(const _Float16* __restrict__ Q,
                            const _Float16* __restrict__ Kt,
                            const _Float16* __restrict__ V,
                            _Float16* __restrict__ O,
                            int Bn, int L, int D)
{
    const int qb   = blockIdx.x * 16;
    const int head = blockIdx.y;
    const int b    = blockIdx.z;
    const int lane = threadIdx.x;
    const int lm   = lane & 15;
    const int lh   = lane >> 4;
    const int hbase = head * 64;
    const int MT = Bn * L;
    const int T  = L / 32;

    __shared__ __align__(16) _Float16 Qs[16 * 64];       // [m][d]
    __shared__ __align__(16) _Float16 Ks[2][32 * 64];    // [key][d]
    __shared__ __align__(16) _Float16 Vs[2][64 * 32];    // [d][key]
    __shared__ __align__(16) _Float16 Ps[16 * 32];       // [m][kcol]

    uint4 vreg[8];
    auto loadVreg = [&](int kb) {
        #pragma unroll
        for (int i = 0; i < 8; ++i) {
            int idx = lane + i * 32;
            int key = idx >> 3, c8 = (idx & 7) * 8;
            vreg[i] = *reinterpret_cast<const uint4*>(
                &V[(size_t)(b * L + kb + key) * D + hbase + c8]);
        }
    };

#if defined(HAVE_TDM)
    tdm_load_2d_f16((unsigned)(size_t)(void*)&Ks[0][0],
                    Kt + (size_t)(b * L) * D + hbase, 64, 32,
                    (unsigned)(D - hbase), (unsigned)MT, (unsigned)D);
    tdm_load_2d_f16((unsigned)(size_t)(void*)&Qs[0],
                    Q + (size_t)(b * L + qb) * D + hbase, 64, 16,
                    (unsigned)(D - hbase), (unsigned)(MT - (b * L + qb)), (unsigned)D);
    __builtin_amdgcn_s_wait_tensorcnt(0);
#else
    #pragma unroll
    for (int i = 0; i < 4; ++i) {
        int idx = lane + i * 32;
        int m = idx >> 3, c8 = (idx & 7) * 8;
        *reinterpret_cast<uint4*>(&Qs[m * 64 + c8]) =
            *reinterpret_cast<const uint4*>(&Q[(size_t)(b * L + qb + m) * D + hbase + c8]);
    }
    #pragma unroll
    for (int i = 0; i < 8; ++i) {
        int idx = lane + i * 32;
        int key = idx >> 3, c8 = (idx & 7) * 8;
        *reinterpret_cast<uint4*>(&Ks[0][key * 64 + c8]) =
            *reinterpret_cast<const uint4*>(&Kt[(size_t)(b * L + key) * D + hbase + c8]);
    }
#endif
    loadVreg(0);
    __syncthreads();

    FragH qa[2];
    #pragma unroll
    for (int t = 0; t < 2; ++t)
        #pragma unroll
        for (int v = 0; v < 8; ++v)
            qa[t].u[v] = *reinterpret_cast<const unsigned int*>(
                &Qs[lm * 64 + t * 32 + frag_kk(v, lh)]);

    v8f oacc[4];
    float mrun[8], lrun[8];
    #pragma unroll
    for (int j = 0; j < 4; ++j)
        #pragma unroll
        for (int r = 0; r < 8; ++r) oacc[j][r] = 0.0f;
    #pragma unroll
    for (int r = 0; r < 8; ++r) { mrun[r] = -1e30f; lrun[r] = 0.0f; }

    const float scale = 0.125f;

    for (int it = 0; it < T; ++it) {
        const int pb = it & 1;

        // commit prefetched V into Vs[pb] (transpose scatter)
        #pragma unroll
        for (int i = 0; i < 8; ++i) {
            int idx = lane + i * 32;
            int key = idx >> 3, c8 = (idx & 7) * 8;
            H8 hv; hv.u = vreg[i];
            #pragma unroll
            for (int t = 0; t < 8; ++t)
                Vs[pb][(c8 + t) * 32 + key] = hv.h[t];
        }

        if (it + 1 < T) {
            const int kb1 = (it + 1) * 32;
#if defined(HAVE_TDM)
            asm volatile("s_wait_dscnt 0x0" ::: "memory");
            tdm_load_2d_f16((unsigned)(size_t)(void*)&Ks[pb ^ 1][0],
                            Kt + (size_t)(b * L + kb1) * D + hbase, 64, 32,
                            (unsigned)(D - hbase), (unsigned)(MT - (b * L + kb1)), (unsigned)D);
#else
            #pragma unroll
            for (int i = 0; i < 8; ++i) {
                int idx = lane + i * 32;
                int key = idx >> 3, c8 = (idx & 7) * 8;
                *reinterpret_cast<uint4*>(&Ks[pb ^ 1][key * 64 + c8]) =
                    *reinterpret_cast<const uint4*>(
                        &Kt[(size_t)(b * L + kb1 + key) * D + hbase + c8]);
            }
#endif
            loadVreg(kb1);                 // global loads overlap compute below
#if defined(HAVE_TDM)
            __builtin_amdgcn_s_wait_tensorcnt(1);   // Ks[pb] (older DMA) done
#endif
        }
#if defined(HAVE_TDM)
        else {
            __builtin_amdgcn_s_wait_tensorcnt(0);
        }
#endif
        __syncthreads();

        // S panel: two 16x16 tiles over 32 keys
        v8f s[2];
        #pragma unroll
        for (int nt = 0; nt < 2; ++nt) {
            FragH b0, b1;
            #pragma unroll
            for (int v = 0; v < 8; ++v) {
                int kk = frag_kk(v, lh);
                b0.u[v] = *reinterpret_cast<const unsigned int*>(&Ks[pb][(nt * 16 + lm) * 64 + kk]);
                b1.u[v] = *reinterpret_cast<const unsigned int*>(&Ks[pb][(nt * 16 + lm) * 64 + 32 + kk]);
            }
            v8f z;
            #pragma unroll
            for (int r = 0; r < 8; ++r) z[r] = 0.0f;
            z     = __builtin_amdgcn_wmma_f32_16x16x32_f16(false, qa[0].h, false, b0.h, (short)0, z, false, false);
            s[nt] = __builtin_amdgcn_wmma_f32_16x16x32_f16(false, qa[1].h, false, b1.h, (short)0, z, false, false);
        }

        float mnew[8], alpha[8];
        #pragma unroll
        for (int r = 0; r < 8; ++r) {
            float a0 = s[0][r] * scale, a1 = s[1][r] * scale;
            s[0][r] = a0; s[1][r] = a1;
            float mx = fmaxf(a0, a1);
            mx = fmaxf(mx, __shfl_xor(mx, 1, 32));
            mx = fmaxf(mx, __shfl_xor(mx, 2, 32));
            mx = fmaxf(mx, __shfl_xor(mx, 4, 32));
            mx = fmaxf(mx, __shfl_xor(mx, 8, 32));
            mnew[r]  = fmaxf(mrun[r], mx);
            alpha[r] = __expf(mrun[r] - mnew[r]);
            mrun[r]  = mnew[r];
        }
        #pragma unroll
        for (int r = 0; r < 8; ++r) {
            float p0 = __expf(s[0][r] - mnew[r]);
            float p1 = __expf(s[1][r] - mnew[r]);
            s[0][r] = p0; s[1][r] = p1;
            float rs = p0 + p1;
            rs += __shfl_xor(rs, 1, 32);
            rs += __shfl_xor(rs, 2, 32);
            rs += __shfl_xor(rs, 4, 32);
            rs += __shfl_xor(rs, 8, 32);
            lrun[r] = lrun[r] * alpha[r] + rs;
            #pragma unroll
            for (int j = 0; j < 4; ++j) oacc[j][r] *= alpha[r];
        }

        __syncthreads();
        #pragma unroll
        for (int r = 0; r < 8; ++r) {
            int m = r + lh * 8;
            Ps[m * 32 + lm]      = (_Float16)s[0][r];
            Ps[m * 32 + 16 + lm] = (_Float16)s[1][r];
        }
        __syncthreads();

        FragH pa;
        #pragma unroll
        for (int v = 0; v < 8; ++v)
            pa.u[v] = *reinterpret_cast<const unsigned int*>(&Ps[lm * 32 + frag_kk(v, lh)]);

        #pragma unroll
        for (int j = 0; j < 4; ++j) {
            FragH vb;
            #pragma unroll
            for (int v = 0; v < 8; ++v)
                vb.u[v] = *reinterpret_cast<const unsigned int*>(
                    &Vs[pb][(j * 16 + lm) * 32 + frag_kk(v, lh)]);
            oacc[j] = __builtin_amdgcn_wmma_f32_16x16x32_f16(false, pa.h, false, vb.h, (short)0, oacc[j], false, false);
        }
        __syncthreads();
    }

    #pragma unroll
    for (int j = 0; j < 4; ++j)
        #pragma unroll
        for (int r = 0; r < 8; ++r) {
            int m = r + lh * 8;
            O[((size_t)(b * L + qb + m)) * D + hbase + j * 16 + lm] =
                (_Float16)(oacc[j][r] / lrun[r]);
        }
}

// ---------------------------------------------------------------------------
// Host orchestration
// ---------------------------------------------------------------------------
static inline void launch_gemm(const _Float16* A, const _Float16* Bt,
                               const float* bias, const float* resid,
                               float* C, _Float16* Ch,
                               int M, int N, int Npad, int K, int act, hipStream_t stream)
{
    dim3 grid(Npad / BN, M / BM);
    gemm_wmma_kernel<<<grid, 256, 0, stream>>>(A, Bt, bias, resid, C, Ch, M, N, Npad, K, act);
}

extern "C" void kernel_launch(void* const* d_in, const int* in_sizes, int n_in,
                              void* d_out, int out_size, void* d_ws, size_t ws_size,
                              hipStream_t stream)
{
    (void)in_sizes; (void)n_in; (void)out_size; (void)ws_size;
    const int Bn = 2, L = 2048, D = 768, H = 12;
    const int M  = Bn * L;                       // 4096 token rows

    const float* x    = (const float*)d_in[0];
    const float* sn   = (const float*)d_in[1];
    const float* cs   = (const float*)d_in[2];
    const float* hw1[3] = { (const float*)d_in[3],  (const float*)d_in[7],  (const float*)d_in[11] };
    const float* hb1[3] = { (const float*)d_in[4],  (const float*)d_in[8],  (const float*)d_in[12] };
    const float* hw2[3] = { (const float*)d_in[5],  (const float*)d_in[9],  (const float*)d_in[13] };
    const float* hb2[3] = { (const float*)d_in[6],  (const float*)d_in[10], (const float*)d_in[14] };
    const float* ln1g = (const float*)d_in[15];
    const float* ln1b = (const float*)d_in[16];
    const float* ln2g = (const float*)d_in[17];
    const float* ln2b = (const float*)d_in[18];
    const float* wq = (const float*)d_in[19];
    const float* wk = (const float*)d_in[20];
    const float* wv = (const float*)d_in[21];
    const float* wo = (const float*)d_in[22];
    const float* bq = (const float*)d_in[23];
    const float* bk = (const float*)d_in[24];
    const float* bv = (const float*)d_in[25];
    const float* bo = (const float*)d_in[26];
    const float* mw1 = (const float*)d_in[27];
    const float* mb1 = (const float*)d_in[28];
    const float* mw2 = (const float*)d_in[29];
    const float* mb2 = (const float*)d_in[30];

    const size_t BLD    = (size_t)M * D;
    const size_t BL4D   = (size_t)M * 4 * D;
    const size_t BL1024 = (size_t)M * 1024;

    // fp32 region
    float* ws   = (float*)d_ws;
    float* hA   = ws;
    float* hB   = hA + BLD;
    float* xln  = hB + BLD;
    float* pbuf = xln + BLD;                    // p (hyena); later attn-out fp32 (BLD prefix)

    // f16 region
    _Float16* x16   = (_Float16*)(pbuf + BL1024);  // x f16; reused as ln1 companion
    _Float16* h16   = x16 + BLD;
    _Float16* t4d16 = h16 + BLD;
    _Float16* qkr16 = t4d16 + BL4D;
    _Float16* q16   = qkr16 + BLD;
    _Float16* k16   = q16 + BLD;
    _Float16* v16   = k16 + BLD;
    _Float16* o16   = v16 + BLD;
    _Float16* x216  = o16 + BLD;
    _Float16* wp    = x216 + BLD;

    const int order_cfg[3] = { 2, 3, 4 };
    const int F_cfg[3]     = { 7, 128, 256 };
    const int p_npad[3]    = { 128, 384, 1024 };   // order*F padded to 128

    // f16 transposed weights
    _Float16* hw1t[3]; _Float16* hw2t[3];
    for (int s = 0; s < 3; ++s) { hw1t[s] = wp; wp += (size_t)3072 * 768; }
    for (int s = 0; s < 3; ++s) { hw2t[s] = wp; wp += (size_t)p_npad[s] * 3072; }
    _Float16* wqT = wp; wp += (size_t)768 * 768;
    _Float16* wkT = wp; wp += (size_t)768 * 768;
    _Float16* wvT = wp; wp += (size_t)768 * 768;
    _Float16* woT = wp; wp += (size_t)768 * 768;
    _Float16* mw1T = wp; wp += (size_t)3072 * 768;
    _Float16* mw2T = wp; wp += (size_t)768 * 3072;

    // ---- pre-pass: convert weights (transposed, padded) and x ----
    auto wconv = [&](const float* W, _Float16* Wt, int K, int N, int Npad) {
        size_t total = (size_t)Npad * K;
        wconvert_kernel<<<dim3((unsigned)((total + 255) / 256)), 256, 0, stream>>>(W, Wt, K, N, Npad);
    };
    for (int s = 0; s < 3; ++s) wconv(hw1[s], hw1t[s], 768, 3072, 3072);
    for (int s = 0; s < 3; ++s) wconv(hw2[s], hw2t[s], 3072, order_cfg[s] * F_cfg[s], p_npad[s]);
    wconv(wq, wqT, 768, 768, 768);
    wconv(wk, wkT, 768, 768, 768);
    wconv(wv, wvT, 768, 768, 768);
    wconv(wo, woT, 768, 768, 768);
    wconv(mw1, mw1T, 768, 3072, 3072);
    wconv(mw2, mw2T, 3072, 768, 768);
    aconvert_kernel<<<dim3((unsigned)((BLD + 255) / 256)), 256, 0, stream>>>(x, x16, BLD);

    // ---- Hyena stages ----
    const float* cur = x;
    const _Float16* curA16 = x16;
    float* pp[2] = { hA, hB };
    int tgt = 0;
    for (int s = 0; s < 3; ++s) {
        int order = order_cfg[s], F = F_cfg[s], orderF = order * F;
        launch_gemm(curA16, hw1t[s], hb1[s], nullptr, nullptr, t4d16, M, 3072, 3072, 768, 1, stream);
        launch_gemm(t4d16, hw2t[s], hb2[s], nullptr, pbuf, nullptr, M, orderF, p_npad[s], 3072, 0, stream);
        for (int i = 0; i < order; ++i) {
            int w16 = (i == order - 1) ? 1 : 0;
            if (F >= 16)
                hyena_conv_tiled_kernel<<<dim3(L / CTL, D / 256, Bn), 256, 0, stream>>>(
                    cur, pbuf, pp[tgt], h16, L, D, F, i, orderF, w16);
            else
                hyena_conv_small_kernel<<<dim3(L, Bn), 256, 0, stream>>>(
                    cur, pbuf, pp[tgt], h16, L, D, F, i, orderF, w16);
            cur = pp[tgt];
            tgt ^= 1;
        }
        curA16 = h16;
    }

    // ---- x = LN(res + h) ----
    ln_res_kernel<<<M, 256, 0, stream>>>(x, cur, ln1g, ln1b, xln, x16, D);  // x16 = f16(xln)
    float* x2_ = (cur == hA) ? hB : hA;

    // ---- RoPE + QKV projections ----
    rope_kernel<<<M, 256, 0, stream>>>(xln, sn, cs, qkr16, L, D);
    launch_gemm(qkr16, wqT, bq, nullptr, nullptr, q16, M, D, D, D, 0, stream);
    launch_gemm(qkr16, wkT, bk, nullptr, nullptr, k16, M, D, D, D, 0, stream);
    launch_gemm(x16,   wvT, bv, nullptr, nullptr, v16, M, D, D, D, 0, stream);

    // ---- attention ----
    attn_kernel<<<dim3(L / 16, H, Bn), 32, 0, stream>>>(q16, k16, v16, o16, Bn, L, D);

    // ---- output projection + LN2 ----
    launch_gemm(o16, woT, bo, nullptr, pbuf, nullptr, M, D, D, D, 0, stream);
    ln_res_kernel<<<M, 256, 0, stream>>>(xln, pbuf, ln2g, ln2b, x2_, x216, D);

    // ---- MLP with fused residual ----
    launch_gemm(x216, mw1T, mb1, nullptr, nullptr, t4d16, M, 3072, 3072, 768, 1, stream);
    launch_gemm(t4d16, mw2T, mb2, x2_, (float*)d_out, nullptr, M, D, D, 3072, 0, stream);
}
